// LinkPredict_87926570484533
// MI455X (gfx1250) — matrix operand order repaired and checked by reading, not compiled, not run on hardware
//
#include <hip/hip_runtime.h>

// ---------------------------------------------------------------------------
// RGCN layer + DistMult scoring for MI455X (gfx1250, wave32, WMMA).
// Pipeline:
//   feats -> fp16
//   pack all weight matrices into WMMA B-fragment order (fp16)
//   x = feats @ W_affine + b            (WMMA f32<-f16, also kept as fp16)
//   for r in 0..7:  tmpf = x @ W_fwd[r]; tmpb = x @ W_bwd[r]
//                   scatter edges of relation r into h via atomic f32 adds
//   h += x @ W_loop                      (WMMA, C initialized from h)
//   weights[t] = sum(h[s]*w_rel[r]*h[o]) (wave-per-triple reduction)
// ---------------------------------------------------------------------------

namespace {
constexpr int kN = 30000;   // nodes  (30000 = 1875 * 16, no ragged M tiles)
constexpr int kE = 300000;  // edges
constexpr int kT = 200000;  // triples
constexpr int kR = 8;       // relations
constexpr int kF = 128;     // raw feature dim
constexpr int kD = 256;     // hidden dim
constexpr int kNT = kD / 16; // 16 column tiles of 16
}

typedef __attribute__((ext_vector_type(16))) _Float16 v16h;
typedef __attribute__((ext_vector_type(8)))  _Float16 v8h;
typedef __attribute__((ext_vector_type(8)))  float    v8f;
typedef __attribute__((ext_vector_type(4)))  int      v4i;

// ---------------------------------------------------------------------------
__global__ __launch_bounds__(256) void cvt_f32_f16(const float* __restrict__ in,
                                                   _Float16* __restrict__ out,
                                                   int n) {
  int i = blockIdx.x * 256 + threadIdx.x;
  if (i < n) out[i] = (_Float16)in[i];
}

__global__ __launch_bounds__(256) void zero_f32(float* __restrict__ p, int n4) {
  int i = blockIdx.x * 256 + threadIdx.x;
  if (i < n4) ((float4*)p)[i] = make_float4(0.f, 0.f, 0.f, 0.f);
}

// Pack a row-major [Kdim x kD] fp32 weight into WMMA B-fragment order (fp16).
// Fragment layout (contiguous k order used here, 16-bit B 32x16): lane (hi=lane>>4, n=lane&15),
// half c in the fragment holds B[kt*32 + hi*16 + c][nt*16 + n].
// Packed index: ((kt*kNT + nt)*32 + lane)*16 + c  -> each lane's fragment is
// 32 contiguous bytes (two b128 loads).
__global__ __launch_bounds__(256) void pack_b(const float* __restrict__ W,
                                              _Float16* __restrict__ out,
                                              int Kdim, long wStride, long oStride) {
  long mat = blockIdx.y;
  int idx = blockIdx.x * 256 + threadIdx.x;
  if (idx >= Kdim * kD) return;
  int c    = idx & 15;
  int lane = (idx >> 4) & 31;
  int blk  = idx >> 9;
  int nt   = blk % kNT;
  int kt   = blk / kNT;
  int hi   = lane >> 4;
  int nn   = lane & 15;
  int krow = kt * 32 + hi * 16 + c;
  int col  = nt * 16 + nn;
  out[mat * oStride + idx] = (_Float16)W[mat * wStride + (long)krow * kD + col];
}

// ---------------------------------------------------------------------------
// WMMA GEMM: C[M x kD] = A[M x K] * B[K x kD]  (K = KCH*128)
// Block = 256 threads = 8 waves, each wave owns a 16-row strip and all 16
// column tiles (acc = 16 x v8f).  B panel staged through LDS in 64KB chunks,
// via the CDNA5 async global->LDS DMA path (ASYNCcnt) when available.
template <int KCH>
__global__ __launch_bounds__(256)
void gemm_wmma(const _Float16* __restrict__ A,     // [M x K] row-major fp16
               const _Float16* __restrict__ Bp,    // packed B fragments
               float* __restrict__ Cout,           // fp32 out (or null)
               const float* __restrict__ Cadd,     // init acc from here (or null)
               const float* __restrict__ bias,     // [kD] col bias (or null)
               _Float16* __restrict__ C16,         // fp16 copy of out (or null)
               int M) {
  __shared__ __align__(16) _Float16 Bs[4 * kNT * 32 * 16];  // 64 KB chunk

  const int lane  = threadIdx.x & 31;
  const int hi    = lane >> 4;
  const int ln    = lane & 15;
  const int mtile = blockIdx.x * 8 + (threadIdx.x >> 5);
  const bool active = (mtile * 16 < M);

  v8f acc[kNT];
#pragma unroll
  for (int nt = 0; nt < kNT; ++nt)
#pragma unroll
    for (int j = 0; j < 8; ++j) acc[nt][j] = 0.f;

  if (active && Cadd) {
#pragma unroll
    for (int nt = 0; nt < kNT; ++nt)
#pragma unroll
      for (int j = 0; j < 8; ++j)
        acc[nt][j] = Cadd[(long)(mtile * 16 + hi * 8 + j) * kD + nt * 16 + ln];
  }

  const _Float16* Arow = A + (long)(mtile * 16 + ln) * (KCH * 128);

  for (int kc = 0; kc < KCH; ++kc) {
    __syncthreads();
    {   // stage this K-chunk's packed B fragments (64 KB) into LDS
      const _Float16* gsrc = Bp + (long)kc * (4 * kNT * 512);
#if defined(__HIP_DEVICE_COMPILE__) && \
    __has_builtin(__builtin_amdgcn_global_load_async_to_lds_b128)
      // gfx1250 async DMA: global -> LDS with no VGPR round-trip (ASYNCcnt).
      for (int i = threadIdx.x; i < 4096; i += 256) {
        __builtin_amdgcn_global_load_async_to_lds_b128(
            (__attribute__((address_space(1))) v4i*)(gsrc + (long)i * 8),
            (__attribute__((address_space(3))) v4i*)(Bs + (long)i * 8),
            /*offset=*/0, /*cpol=*/0);
      }
      asm volatile("s_wait_asynccnt 0x0" ::: "memory");
#else
      const uint4* g = (const uint4*)gsrc;
      uint4* l = (uint4*)Bs;
      for (int i = threadIdx.x; i < 4096; i += 256) l[i] = g[i];
#endif
    }
    __syncthreads();

    if (active) {
      // A fragments for the 4 K-tiles of this chunk.
      // Lane needs halves K = [kt*32 + hi*16, +16) (contiguous, matches packed B):
      // two contiguous 16B global loads per fragment.
      v16h afrag[4];
#pragma unroll
      for (int kt = 0; kt < 4; ++kt) {
        const _Float16* ap = Arow + (kc * 4 + kt) * 32;
        v8h a0 = *(const v8h*)(ap + hi * 16);
        v8h a1 = *(const v8h*)(ap + hi * 16 + 8);
        afrag[kt] = __builtin_shufflevector(a0, a1,
            0, 1, 2, 3, 4, 5, 6, 7, 8, 9, 10, 11, 12, 13, 14, 15);
      }
#pragma unroll
      for (int nt = 0; nt < kNT; ++nt) {
#pragma unroll
        for (int kt = 0; kt < 4; ++kt) {
          const _Float16* bp = Bs + ((kt * kNT + nt) * 32 + lane) * 16;
          v8h b0 = *(const v8h*)bp;         // ds_load_b128
          v8h b1 = *(const v8h*)(bp + 8);   // ds_load_b128
          v16h bfr = __builtin_shufflevector(b0, b1,
              0, 1, 2, 3, 4, 5, 6, 7, 8, 9, 10, 11, 12, 13, 14, 15);
          acc[nt] = __builtin_amdgcn_wmma_f32_16x16x32_f16(
              false, afrag[kt], false, bfr, (short)0, acc[nt], false, false);
        }
      }
    }
  }

  if (!active) return;
#pragma unroll
  for (int nt = 0; nt < kNT; ++nt) {
    float bv = bias ? bias[nt * 16 + ln] : 0.f;
#pragma unroll
    for (int j = 0; j < 8; ++j) {
      float v = acc[nt][j] + bv;
      long off = (long)(mtile * 16 + hi * 8 + j) * kD + nt * 16 + ln;
      if (Cout) Cout[off] = v;
      if (C16)  C16[off]  = (_Float16)v;
    }
  }
}

// ---------------------------------------------------------------------------
// Scatter edges of one relation: 64 threads per edge, float4 per thread.
// fwd: h[dst] += tmpf[src];  bwd: h[src] += tmpb[dst]
__global__ __launch_bounds__(256)
void scatter_rel(const int* __restrict__ src, const int* __restrict__ dst,
                 const int* __restrict__ etype,
                 const float* __restrict__ tf, const float* __restrict__ tb,
                 float* __restrict__ h, int rel, int E) {
  int idx = blockIdx.x * 256 + threadIdx.x;
  int e = idx >> 6;
  if (e >= E || etype[e] != rel) return;
  int c = (idx & 63) * 4;
  int s = src[e], d = dst[e];
  float4 mf = *(const float4*)(tf + (long)s * kD + c);
  float4 mb = *(const float4*)(tb + (long)d * kD + c);
  float* hd = h + (long)d * kD + c;
  float* hs = h + (long)s * kD + c;
  atomicAdd(hd + 0, mf.x); atomicAdd(hd + 1, mf.y);
  atomicAdd(hd + 2, mf.z); atomicAdd(hd + 3, mf.w);
  atomicAdd(hs + 0, mb.x); atomicAdd(hs + 1, mb.y);
  atomicAdd(hs + 2, mb.z); atomicAdd(hs + 3, mb.w);
}

// ---------------------------------------------------------------------------
// DistMult: one wave32 per triple, 8 floats per lane, shuffle-tree reduce.
__global__ __launch_bounds__(256)
void distmult(const float* __restrict__ h, const float* __restrict__ w_rel,
              const int* __restrict__ ts, const int* __restrict__ tr,
              const int* __restrict__ to, float* __restrict__ out, int T) {
  int g = (blockIdx.x * 256 + threadIdx.x) >> 5;
  if (g >= T) return;
  int lane = threadIdx.x & 31;
  const float4* hs = (const float4*)(h + (long)ts[g] * kD + lane * 8);
  const float4* ho = (const float4*)(h + (long)to[g] * kD + lane * 8);
  const float4* hr = (const float4*)(w_rel + (long)tr[g] * kD + lane * 8);
  float sum = 0.f;
#pragma unroll
  for (int q = 0; q < 2; ++q) {
    float4 a = hs[q], b = hr[q], c = ho[q];
    sum += a.x * b.x * c.x + a.y * b.y * c.y + a.z * b.z * c.z + a.w * b.w * c.w;
  }
#pragma unroll
  for (int off = 16; off > 0; off >>= 1)
    sum += __shfl_xor(sum, off, 32);
  if (lane == 0) out[g] = sum;
}

// ---------------------------------------------------------------------------
extern "C" void kernel_launch(void* const* d_in, const int* in_sizes, int n_in,
                              void* d_out, int out_size, void* d_ws, size_t ws_size,
                              hipStream_t stream) {
  (void)in_sizes; (void)n_in; (void)out_size; (void)ws_size;

  const float* feats  = (const float*)d_in[0];
  const float* W_aff  = (const float*)d_in[1];
  const float* b_aff  = (const float*)d_in[2];
  const float* W_fwd  = (const float*)d_in[3];
  const float* W_bwd  = (const float*)d_in[4];
  const float* W_loop = (const float*)d_in[5];
  const float* w_rel  = (const float*)d_in[6];
  const int* src   = (const int*)d_in[7];
  const int* dst   = (const int*)d_in[8];
  const int* etype = (const int*)d_in[9];
  const int* tr_s  = (const int*)d_in[10];
  const int* tr_r  = (const int*)d_in[11];
  const int* tr_o  = (const int*)d_in[12];

  float* out_w = (float*)d_out;     // [T] triple scores
  float* h     = out_w + kT;        // [N x D] node embeddings (also agg buffer)

  // Workspace carve-up (all offsets 16B aligned); ~85 MB total.
  char* ws = (char*)d_ws;
  _Float16* feats16 = (_Float16*)ws;                       //  7.68 MB
  _Float16* xh      = feats16 + (long)kN * kF;             // 15.36 MB
  float* tmpf       = (float*)(xh + (long)kN * kD);        // 30.72 MB
  float* tmpb       = tmpf + (long)kN * kD;                // 30.72 MB
  _Float16* pwA     = (_Float16*)(tmpb + (long)kN * kD);   // 64 KB
  _Float16* pwF     = pwA + kF * kD;                       // 1 MB
  _Float16* pwB     = pwF + (long)kR * kD * kD;            // 1 MB
  _Float16* pwL     = pwB + (long)kR * kD * kD;            // 128 KB

  // --- preprocessing: fp16 feats + packed fp16 weight fragments
  cvt_f32_f16<<<(kN * kF + 255) / 256, 256, 0, stream>>>(feats, feats16, kN * kF);
  pack_b<<<dim3((kF * kD + 255) / 256, 1), 256, 0, stream>>>(W_aff, pwA, kF, 0, 0);
  pack_b<<<dim3((kD * kD + 255) / 256, kR), 256, 0, stream>>>(
      W_fwd, pwF, kD, (long)kD * kD, (long)kD * kD);
  pack_b<<<dim3((kD * kD + 255) / 256, kR), 256, 0, stream>>>(
      W_bwd, pwB, kD, (long)kD * kD, (long)kD * kD);
  pack_b<<<dim3((kD * kD + 255) / 256, 1), 256, 0, stream>>>(W_loop, pwL, kD, 0, 0);
  zero_f32<<<(kN * kD / 4 + 255) / 256, 256, 0, stream>>>(h, kN * kD / 4);

  const int gemmBlocks = (kN / 16 + 7) / 8;  // 235 blocks of 8 waves

  // --- x = feats @ W_affine + b  (kept only as fp16 for downstream GEMMs)
  gemm_wmma<1><<<gemmBlocks, 256, 0, stream>>>(
      feats16, pwA, nullptr, nullptr, b_aff, xh, kN);

  // --- per-relation transform + edge scatter (tmp buffers are L2-resident)
  for (int r = 0; r < kR; ++r) {
    gemm_wmma<2><<<gemmBlocks, 256, 0, stream>>>(
        xh, pwF + (long)r * kD * kD, tmpf, nullptr, nullptr, nullptr, kN);
    gemm_wmma<2><<<gemmBlocks, 256, 0, stream>>>(
        xh, pwB + (long)r * kD * kD, tmpb, nullptr, nullptr, nullptr, kN);
    scatter_rel<<<(kE * 64 + 255) / 256, 256, 0, stream>>>(
        src, dst, etype, tmpf, tmpb, h, r, kE);
  }

  // --- self loop: h = agg + x @ W_loop (acc initialized from aggregated h)
  gemm_wmma<2><<<gemmBlocks, 256, 0, stream>>>(
      xh, pwL, h, h, nullptr, nullptr, kN);

  // --- DistMult scoring
  distmult<<<(kT * 32 + 255) / 256, 256, 0, stream>>>(
      h, w_rel, tr_s, tr_r, tr_o, out_w, kT);
}
